// VectorQuantizer_42339787604544
// MI455X (gfx1250) — matrix operand order repaired and checked by reading, not compiled, not run on hardware
//
#include <hip/hip_runtime.h>
#include <hip/hip_bf16.h>

// ---------------- problem constants ----------------
#define CQ   256
#define THW  8192           // T*H*W = 8*32*32
#define KQ   1024
#define NQ   65536          // B*T*H*W rows
#define NCQ  16777216       // N*C elements in output tensor
#define KTILES (KQ/16)      // 64 code tiles
#define CHUNKS (CQ/32)      // 8 K-chunks per WMMA row
#define WAVES 8             // waves per block (wave32) -> 128 rows / block
#define BTILE_US 4096       // ushorts per packed B code-tile (8 KB)

typedef __attribute__((ext_vector_type(16))) __bf16 v16bf;
typedef __attribute__((ext_vector_type(8)))  float  v8f;
typedef __attribute__((ext_vector_type(4)))  unsigned int u32x4;
typedef __attribute__((ext_vector_type(8)))  int i32x8;
typedef __attribute__((ext_vector_type(4)))  int i32x4;

union AFrag { v16bf v; unsigned int d[8]; };
union BFrag { v16bf v; unsigned int d[8]; };

static __device__ __forceinline__ unsigned short f2bf(float f) {
    unsigned int u = __float_as_uint(f);
    unsigned int r = (u + 0x7FFFu + ((u >> 16) & 1u)) >> 16;   // RNE
    return (unsigned short)r;
}

// ---------------------------------------------------------------------------
// TDM: 1-D copy of one packed B code-tile (8 KB = 2048 dwords) global -> LDS.
// D# group0: count=1 | lds_addr | global_addr[56:0] | type=2 ("image")
// D# group1: data_size=2 (4B) | tensor_dim0=2048 | tensor_dim1=1 |
//            tile_dim0=2048 | tile_dim1=0 (1-D) | dim0_stride=2048
// 6-arg builtin form (clang-23 / therock-10.0 lane).
// ---------------------------------------------------------------------------
static __device__ __forceinline__ void tdm_load_b_tile(const unsigned short* gsrc,
                                                       unsigned int lds_off) {
    unsigned long long ga = (unsigned long long)(uintptr_t)gsrc;
    u32x4 g0;
    g0.x = 0x1u;                                          // count=1, user D#
    g0.y = lds_off;                                       // lds_addr (bytes)
    g0.z = (unsigned int)(ga & 0xFFFFFFFFu);              // global_addr[31:0]
    g0.w = (unsigned int)((ga >> 32) & 0x01FFFFFFu)       // global_addr[56:32]
         | (2u << 30);                                    // type=2
    i32x8 g1;
    g1[0] = (int)(2u << 16);            // workgroup_mask=0, data_size=4B
    g1[1] = (int)((2048u & 0xFFFFu) << 16);   // tensor_dim0[15:0] @ bits[63:48]
    g1[2] = (int)(1u << 16);            // tensor_dim0[31:16]=0 | tensor_dim1[15:0]=1
    g1[3] = (int)(2048u << 16);         // tensor_dim1[31:16]=0 | tile_dim0=2048
    g1[4] = 0;                          // tile_dim1=0 (1-D), tile_dim2=0
    g1[5] = 2048;                       // tensor_dim0_stride[31:0]
    g1[6] = 0;
    g1[7] = 0;
    i32x4 z4 = {};
    i32x8 z8 = {};
    __builtin_amdgcn_tensor_load_to_lds(g0, g1, z4, z4, z8, 0); // TENSORcnt++
}

// ---------------------------------------------------------------------------
// Kernel 1: pack codebook into B-fragment order (bf16), half-norms, zero loss.
// B fragment layout (V_WMMA_*_16x16x32 bf16, wave32, ISA 7.12.2):
//   element e (0..15) of lane l -> B[K = (e>=8?16:0)+(l>=16?8:0)+(e&7)][N = l&15]
//   B[K][N] = codebook[code = kt*16+N][chan = chunk*32+K]
// packB flat index: (((kt*CHUNKS + chunk)*32 + lane)*16 + e)
// ---------------------------------------------------------------------------
__global__ void vq_pack_kernel(const float* __restrict__ cb,
                               unsigned short* __restrict__ packB,
                               float* __restrict__ halfnorm,
                               float* __restrict__ lossAcc) {
    int tid = blockIdx.x * blockDim.x + threadIdx.x;
    if (tid == 0) *lossAcc = 0.0f;

    if (tid < KTILES * CHUNKS * 32 * 16) {
        int e    = tid & 15;
        int lane = (tid >> 4) & 31;
        int ch   = (tid >> 9) & (CHUNKS - 1);
        int kt   = tid >> 12;
        int code = kt * 16 + (lane & 15);
        int chan = ch * 32 + ((e >> 3) << 4) + ((lane >> 4) << 3) + (e & 7);
        packB[tid] = f2bf(cb[code * CQ + chan]);
    }
    if (tid < KQ) {
        float s = 0.0f;
        const float* row = cb + tid * CQ;
        #pragma unroll 4
        for (int c = 0; c < CQ; ++c) { float v = row[c]; s += v * v; }
        halfnorm[tid] = 0.5f * s;
    }
}

// ---------------------------------------------------------------------------
// Kernel 2: per-wave 16-row tile; B double-buffered in LDS via TDM;
// two independent WMMA accumulation chains; argmax; fp32 gather + loss.
// score = x.c - 0.5*||c||^2  (argmax score == argmin squared-L2 distance)
// ---------------------------------------------------------------------------
__global__ __launch_bounds__(WAVES * 32)
void vq_main_kernel(const float* __restrict__ x,
                    const float* __restrict__ cb,
                    const unsigned short* __restrict__ packB,
                    const float* __restrict__ halfnorm,
                    float* __restrict__ out,
                    float* __restrict__ lossAcc) {
    __shared__ unsigned short bufB[2][BTILE_US];   // 2 x 8 KB double buffer
    __shared__ int            wIdx[WAVES][16];

    const int lane = threadIdx.x & 31;
    const int wave = threadIdx.x >> 5;
    const int tile = blockIdx.x * WAVES + wave;     // 16-row tile id
    const int n0   = tile * 16;
    const int bq   = n0 >> 13;                      // / THW (tiles never cross b)
    const int thw0 = n0 & (THW - 1);

    const float* xb = x + (size_t)bq * CQ * THW + thw0;

    // ---- build 8 A fragments directly from global x (ISA A 16x32 layout) ----
    // dword j of lane l holds channels Kb,Kb+1: Kb=(j<4?2j:2j+8)+(l>=16?8:0),
    // row M=l&15.  Half-wave loads are 64B-contiguous over the 16 rows.
    const int m   = lane & 15;
    const int hi8 = (lane >> 4) << 3;
    AFrag a[CHUNKS];
    #pragma unroll
    for (int ch = 0; ch < CHUNKS; ++ch) {
        #pragma unroll
        for (int j = 0; j < 8; ++j) {
            int Kb = (j < 4 ? 2 * j : 2 * j + 8) + hi8;
            int c  = ch * 32 + Kb;
            float f0 = xb[(size_t)c * THW + m];
            float f1 = xb[(size_t)(c + 1) * THW + m];
            a[ch].d[j] = (unsigned int)f2bf(f0) | ((unsigned int)f2bf(f1) << 16);
        }
    }

    // ---- prologue: TDM-stage code tile 0 into buffer 0 ----
    const unsigned int ldsOff[2] = {
        (unsigned int)(uintptr_t)&bufB[0][0],
        (unsigned int)(uintptr_t)&bufB[1][0]
    };
    if (wave == 0) {
        tdm_load_b_tile(packB, ldsOff[0]);
        __builtin_amdgcn_s_wait_tensorcnt(0);
    }
    __syncthreads();

    float best[8];
    int   bidx[8];
    #pragma unroll
    for (int r = 0; r < 8; ++r) { best[r] = -3.0e38f; bidx[r] = 0; }
    const int col = lane & 15;                      // N column this lane scores

    for (int kt = 0; kt < KTILES; ++kt) {
        const int cur = kt & 1;
        // async TDM fetch of next code tile into the other buffer
        if (wave == 0 && kt + 1 < KTILES)
            tdm_load_b_tile(packB + (size_t)(kt + 1) * BTILE_US, ldsOff[cur ^ 1]);

        // two independent WMMA accumulation chains over the 8 K-chunks
        const unsigned short* ldsB = &bufB[cur][0];
        v8f acc0 = {}, acc1 = {};
        #pragma unroll
        for (int ch = 0; ch < 4; ++ch) {
            BFrag b0, b1;
            b0.v = *(const v16bf*)(ldsB + (ch    ) * 512 + lane * 16);
            b1.v = *(const v16bf*)(ldsB + (ch + 4) * 512 + lane * 16);
            acc0 = __builtin_amdgcn_wmma_f32_16x16x32_bf16(
                       false, a[ch].v,     false, b0.v, (short)0, acc0, false, false);
            acc1 = __builtin_amdgcn_wmma_f32_16x16x32_bf16(
                       false, a[ch + 4].v, false, b1.v, (short)0, acc1, false, false);
        }

        const float hc   = halfnorm[kt * 16 + col];
        const int   code = kt * 16 + col;
        #pragma unroll
        for (int r = 0; r < 8; ++r) {
            float s = (acc0[r] + acc1[r]) - hc;
            if (s > best[r]) { best[r] = s; bidx[r] = code; }
        }

        // buffer handoff: producer drains TENSORcnt, then block barrier
        if (wave == 0) __builtin_amdgcn_s_wait_tensorcnt(0);
        __syncthreads();
    }

    // ---- argmax reduction across the 16 columns of each half-wave ----
    // C/D layout: VGPR r, lane l -> M = r + 8*(l>=16), N = l&15
    #pragma unroll
    for (int r = 0; r < 8; ++r) {
        float s = best[r]; int i = bidx[r];
        #pragma unroll
        for (int off = 8; off > 0; off >>= 1) {
            float so = __shfl_xor(s, off, 16);
            int   io = __shfl_xor(i, off, 16);
            if (so > s || (so == s && io < i)) { s = so; i = io; }
        }
        if ((lane & 15) == 0) wIdx[wave][r + ((lane >> 4) << 3)] = i;
    }
    __syncthreads();

    // ---- output gather + loss (fp32, 64B-coalesced 16-row runs) ----
    const int  code = wIdx[wave][m];
    const float* cbrow = cb + (size_t)code * CQ;
    float* ob = out + (size_t)bq * CQ * THW + thw0 + m;
    const float* xr = xb + m;
    float lsum = 0.0f;
    for (int c = (lane >> 4); c < CQ; c += 2) {
        size_t off = (size_t)c * THW;
        float q  = cbrow[c];
        float xv = xr[off];
        ob[off] = q;
        float d = q - xv;
        lsum += d * d;
    }
    #pragma unroll
    for (int off = 16; off > 0; off >>= 1)
        lsum += __shfl_xor(lsum, off, 32);
    if (lane == 0) atomicAdd(lossAcc, lsum);
}

// ---------------------------------------------------------------------------
// Kernel 3: loss = 1.25 * sum / (N*C)
// ---------------------------------------------------------------------------
__global__ void vq_finalize_kernel(const float* __restrict__ lossAcc,
                                   float* __restrict__ out) {
    if (threadIdx.x == 0 && blockIdx.x == 0)
        out[NCQ] = 1.25f * (*lossAcc) * (1.0f / (float)NCQ);
}

// ---------------------------------------------------------------------------
extern "C" void kernel_launch(void* const* d_in, const int* in_sizes, int n_in,
                              void* d_out, int out_size, void* d_ws, size_t ws_size,
                              hipStream_t stream) {
    const float* x  = (const float*)d_in[0];   // [8,256,8,32,32] fp32
    const float* cb = (const float*)d_in[1];   // [1024,256] fp32
    float* out = (float*)d_out;                // [8,256,8,32,32] + loss scalar

    // workspace layout
    unsigned short* packB    = (unsigned short*)d_ws;                    // 512 KiB
    float*          halfnorm = (float*)((char*)d_ws + 524288);           // 4 KiB
    float*          lossAcc  = (float*)((char*)d_ws + 524288 + 4096);    // 4 B

    // 1) pack codebook (bf16 B-fragments) + half-norms + zero loss
    vq_pack_kernel<<<(KTILES * CHUNKS * 32 * 16) / 256, 256, 0, stream>>>(
        cb, packB, halfnorm, lossAcc);

    // 2) main: 65536 rows / (8 waves * 16 rows) = 512 blocks of 256 threads
    vq_main_kernel<<<NQ / (WAVES * 16), WAVES * 32, 0, stream>>>(
        x, cb, packB, halfnorm, out, lossAcc);

    // 3) finalize scalar loss
    vq_finalize_kernel<<<1, 32, 0, stream>>>(lossAcc, out);
}
